// MultiHeadAttention_5952824673210
// MI455X (gfx1250) — matrix operand order repaired and checked by reading, not compiled, not run on hardware
//
#include <hip/hip_runtime.h>
#include <math.h>

// ---- CDNA5 (gfx1250) wave32 WMMA types -------------------------------------
typedef __bf16 bf16;
typedef __attribute__((ext_vector_type(16))) __bf16 v16bf;
typedef __attribute__((ext_vector_type(8)))  __bf16 v8bf;
typedef __attribute__((ext_vector_type(4)))  __bf16 v4bf;
typedef __attribute__((ext_vector_type(8)))  float  v8f;

#define WMMA_BF16(a, b, c) \
  __builtin_amdgcn_wmma_f32_16x16x32_bf16(false, (a), false, (b), (short)0, (c), false, false)

#define DM   512   // d_model
#define NH   8     // heads
#define DH   64    // head dim
#define NSEQ 2048  // sequence length
#define NB   8     // batch

// ---- DPP16 butterfly reductions within each 16-lane row (no LDS, no waits) --
// xor1 = quad_perm(1,0,3,2)=0xB1, xor2 = quad_perm(2,3,0,1)=0x4E,
// xor4 = ROW_HALF_MIRROR=0x141,   xor8 = ROW_MIRROR=0x140
#define DPP_F32(v, ctrl) \
  __int_as_float(__builtin_amdgcn_update_dpp(0, __float_as_int(v), ctrl, 0xF, 0xF, true))

__device__ __forceinline__ float row_sum16(float v) {
  v += DPP_F32(v, 0xB1);
  v += DPP_F32(v, 0x4E);
  v += DPP_F32(v, 0x141);
  v += DPP_F32(v, 0x140);
  return v;
}
__device__ __forceinline__ float row_max16(float v) {
  v = fmaxf(v, DPP_F32(v, 0xB1));
  v = fmaxf(v, DPP_F32(v, 0x4E));
  v = fmaxf(v, DPP_F32(v, 0x141));
  v = fmaxf(v, DPP_F32(v, 0x140));
  return v;
}

// ---- fragment loaders (layouts per ISA 7.12.2, wave32) ---------------------
__device__ __forceinline__ v16bf load_a16(const bf16* base, int lda, int lane) {
  int m  = lane & 15;
  int kb = (lane >> 4) << 3;           // 0 or 8
  const bf16* p = base + m * lda + kb;
  v8bf lo = *(const v8bf*)(p);
  v8bf hi = *(const v8bf*)(p + 16);
  return __builtin_shufflevector(lo, hi, 0,1,2,3,4,5,6,7,8,9,10,11,12,13,14,15);
}

__device__ __forceinline__ v16bf load_b16(const bf16* base, int ldb, int lane) {
  int n  = lane & 15;
  int kb = (lane >> 4) << 3;           // 0 or 8
  const bf16* p = base + (size_t)n * ldb + kb;
  v8bf lo = *(const v8bf*)(p);
  v8bf hi = *(const v8bf*)(p + 16);
  return __builtin_shufflevector(lo, hi, 0,1,2,3,4,5,6,7,8,9,10,11,12,13,14,15);
}

// ============================================================================
// Kernel 0: one-time weight prep: W[k][n] f32 -> Wt[n][k] bf16 (LDS-tiled).
// ============================================================================
__global__ __launch_bounds__(256) void wtrans_kernel(
    const float* __restrict__ w0, const float* __restrict__ w1,
    const float* __restrict__ w2, const float* __restrict__ w3,
    bf16* __restrict__ o0, bf16* __restrict__ o1,
    bf16* __restrict__ o2, bf16* __restrict__ o3) {
  __shared__ bf16 tile[64][66];
  const float* W = blockIdx.z == 0 ? w0 : blockIdx.z == 1 ? w1
                 : blockIdx.z == 2 ? w2 : w3;
  bf16* O = blockIdx.z == 0 ? o0 : blockIdx.z == 1 ? o1
          : blockIdx.z == 2 ? o2 : o3;
  const int k0 = blockIdx.x * 64, n0 = blockIdx.y * 64;
  const int c = threadIdx.x & 63, rb = threadIdx.x >> 6;
  for (int r = rb; r < 64; r += 4)
    tile[r][c] = (bf16)W[(size_t)(k0 + r) * DM + n0 + c];
  __syncthreads();
  for (int r = rb; r < 64; r += 4)
    O[(size_t)(n0 + r) * DM + k0 + c] = tile[c][r];
}

// ============================================================================
// Kernel 1: Q/K/V projection + shared LayerNorm. 512 thr (16 waves), 64 rows
// per block. B-fragments direct from global bf16 Wt; no barriers in K-loop.
// Q output is pre-scaled by 1/sqrt(Dh) (folds the attention scale away).
// ============================================================================
__global__ __launch_bounds__(512) void qkv_ln_kernel(
    const float* __restrict__ x,  const bf16* __restrict__ wtq,
    const bf16* __restrict__ wtk, const bf16* __restrict__ wtv,
    const float* __restrict__ g,  const float* __restrict__ bta,
    bf16* __restrict__ qo, bf16* __restrict__ ko, bf16* __restrict__ vo) {
  __shared__ bf16  xs[64 * DM];        // 64 KB
  __shared__ float s_sum[64], s_sq[64];

  const int tid  = threadIdx.x;
  const int lane = tid & 31;
  const int wave = tid >> 5;
  const int half = lane >> 4, col = lane & 15;
  const int row0 = blockIdx.x * 64;

  {  // stage x tile, float4 loads -> v4bf stores
    const float4* xv = (const float4*)(x + (size_t)row0 * DM);
    for (int i = tid; i < 64 * DM / 4; i += 512) {
      float4 f = xv[i];
      v4bf b4 = {(bf16)f.x, (bf16)f.y, (bf16)f.z, (bf16)f.w};
      *(v4bf*)(xs + i * 4) = b4;
    }
  }
  __syncthreads();

  const bf16* wlist[3] = {wtq, wtk, wtv};
  bf16*       olist[3] = {qo, ko, vo};

  for (int widx = 0; widx < 3; ++widx) {
    const bf16* WT = wlist[widx];
    v8f acc[4][2];
    #pragma unroll
    for (int m = 0; m < 4; ++m)
      #pragma unroll
      for (int nt = 0; nt < 2; ++nt)
        #pragma unroll
        for (int r = 0; r < 8; ++r) acc[m][nt][r] = 0.f;

    for (int kt = 0; kt < 16; ++kt) {
      v16bf a[4];
      #pragma unroll
      for (int m = 0; m < 4; ++m)
        a[m] = load_a16(xs + (m * 16) * DM + kt * 32, DM, lane);
      #pragma unroll
      for (int nt = 0; nt < 2; ++nt) {
        v16bf bm = load_b16(WT + (size_t)(wave * 32 + nt * 16) * DM + kt * 32,
                            DM, lane);
        #pragma unroll
        for (int m = 0; m < 4; ++m)
          acc[m][nt] = WMMA_BF16(a[m], bm, acc[m][nt]);
      }
    }

    // ---- shared LayerNorm over all 512 columns of the 64 rows ----
    __syncthreads();
    if (tid < 64) { s_sum[tid] = 0.f; s_sq[tid] = 0.f; }
    __syncthreads();
    #pragma unroll
    for (int m = 0; m < 4; ++m)
      #pragma unroll
      for (int r = 0; r < 8; ++r) {
        float s  = row_sum16(acc[m][0][r] + acc[m][1][r]);
        float qq = row_sum16(acc[m][0][r] * acc[m][0][r] +
                             acc[m][1][r] * acc[m][1][r]);
        if (col == 0) {
          atomicAdd(&s_sum[m * 16 + r + half * 8], s);
          atomicAdd(&s_sq [m * 16 + r + half * 8], qq);
        }
      }
    __syncthreads();
    bf16* O = olist[widx];
    const float oscale = (widx == 0) ? 0.125f : 1.0f;  // 1/sqrt(64) into Q
    #pragma unroll
    for (int m = 0; m < 4; ++m)
      #pragma unroll
      for (int r = 0; r < 8; ++r) {
        int   lrow = m * 16 + r + half * 8;
        float mean = s_sum[lrow] * (1.f / DM);
        float var  = s_sq [lrow] * (1.f / DM) - mean * mean;
        float rstd = rsqrtf(var + 1e-6f);
        int   grow = row0 + lrow;
        int   b = grow >> 11, sp = grow & (NSEQ - 1);
        #pragma unroll
        for (int nt = 0; nt < 2; ++nt) {
          int   n = wave * 32 + nt * 16 + col;
          float y = ((acc[m][nt][r] - mean) * rstd * g[n] + bta[n]) * oscale;
          O[(((size_t)b * NH + (n >> 6)) * NSEQ + sp) * DH + (n & 63)] = (bf16)y;
        }
      }
  }
}

// ============================================================================
// Kernel 2: flash attention. grid (NSEQ/128, B*H), 256 thr (8 waves), wave
// owns 16 query rows. 64-key chunks, double-buffered LDS, software-pipelined
// prefetch, 1 barrier/chunk. Keys within a chunk are stored PERMUTED
// (j -> (j%4)*16 + j/4) so softmax P writes are contiguous b64 stores.
// ============================================================================
__global__ __launch_bounds__(256) void attn_kernel(
    const bf16* __restrict__ Q, const bf16* __restrict__ K,
    const bf16* __restrict__ V, bf16* __restrict__ AO) {
  __shared__ bf16 ks[2][64 * DH];       // [slot][d]    2 x 8 KB
  __shared__ bf16 vs[2][DH * 64];       // [d][slot]    2 x 8 KB
  __shared__ bf16 ps[8][16 * 64];       // per-wave P   16 KB

  const int tid  = threadIdx.x;
  const int lane = tid & 31;
  const int wave = tid >> 5;
  const int half = lane >> 4, col = lane & 15;
  const int bh   = blockIdx.y;
  const int q0   = blockIdx.x * 128 + wave * 16;

  const bf16* qp = Q + ((size_t)bh * NSEQ + q0) * DH;
  const bf16* kp = K + (size_t)bh * NSEQ * DH;
  const bf16* vp = V + (size_t)bh * NSEQ * DH;

  v16bf aq0 = load_a16(qp, DH, lane);        // d 0..31 (Q pre-scaled by 1/8)
  v16bf aq1 = load_a16(qp + 32, DH, lane);   // d 32..63

  v8f o[4];
  #pragma unroll
  for (int t = 0; t < 4; ++t)
    #pragma unroll
    for (int r = 0; r < 8; ++r) o[t][r] = 0.f;
  float mrun[8], lrun[8];
  #pragma unroll
  for (int r = 0; r < 8; ++r) { mrun[r] = -1e30f; lrun[r] = 0.f; }

  bf16* pw = ps[wave];
  const int srow = tid >> 2, sd = (tid & 3) * 16;       // staging row / d-off
  const int prow = (srow & 3) * 16 + (srow >> 2);       // permuted key slot

  auto compute = [&](const bf16* ksb, const bf16* vsb) {
    v8f s[4];
    #pragma unroll
    for (int jt = 0; jt < 4; ++jt) {                    // 4 key-slot tiles
      #pragma unroll
      for (int r = 0; r < 8; ++r) s[jt][r] = 0.f;
      s[jt] = WMMA_BF16(aq0, load_b16(ksb + jt * 16 * DH, DH, lane), s[jt]);
      s[jt] = WMMA_BF16(aq1, load_b16(ksb + jt * 16 * DH + 32, DH, lane), s[jt]);
    }
    #pragma unroll
    for (int r = 0; r < 8; ++r) {                       // online softmax, f32
      float e0 = s[0][r], e1 = s[1][r], e2 = s[2][r], e3 = s[3][r];
      float rm = row_max16(fmaxf(fmaxf(e0, e1), fmaxf(e2, e3)));
      float nm = fmaxf(mrun[r], rm);
      float f  = __expf(mrun[r] - nm);
      mrun[r]  = nm;
      float p0 = __expf(e0 - nm), p1 = __expf(e1 - nm);
      float p2 = __expf(e2 - nm), p3 = __expf(e3 - nm);
      float rs = row_sum16((p0 + p1) + (p2 + p3));
      lrun[r] = lrun[r] * f + rs;
      o[0][r] *= f; o[1][r] *= f; o[2][r] *= f; o[3][r] *= f;
      // lane's 4 tile values are keys col*4 + {0,1,2,3}: one contiguous b64
      v4bf pv = {(bf16)p0, (bf16)p1, (bf16)p2, (bf16)p3};
      *(v4bf*)(pw + (r + half * 8) * 64 + col * 4) = pv;
    }
    v16bf ap0 = load_a16(pw, 64, lane);                 // slots 0..31
    v16bf ap1 = load_a16(pw + 32, 64, lane);            // slots 32..63
    #pragma unroll
    for (int t = 0; t < 4; ++t) {
      o[t] = WMMA_BF16(ap0, load_b16(vsb + t * 16 * 64, 64, lane), o[t]);
      o[t] = WMMA_BF16(ap1, load_b16(vsb + t * 16 * 64 + 32, 64, lane), o[t]);
    }
  };

  const int NC = NSEQ / 64;
  {  // stage chunk 0 into buffer 0 (permuted slots)
    size_t off = (size_t)srow * DH + sd;
    v16bf kr = *(const v16bf*)(kp + off);
    v16bf vr = *(const v16bf*)(vp + off);
    *(v16bf*)(&ks[0][prow * DH + sd]) = kr;
    #pragma unroll
    for (int i = 0; i < 16; ++i) vs[0][(sd + i) * 64 + srow] = vr[i];
  }
  __syncthreads();

  for (int c = 0; c < NC - 1; ++c) {
    const int cur = c & 1, nxt = cur ^ 1;
    size_t off = (size_t)((c + 1) * 64 + srow) * DH + sd;
    v16bf kr = *(const v16bf*)(kp + off);               // prefetch in flight
    v16bf vr = *(const v16bf*)(vp + off);
    compute(ks[cur], vs[cur]);                          // ...across WMMA block
    *(v16bf*)(&ks[nxt][prow * DH + sd]) = kr;
    #pragma unroll
    for (int i = 0; i < 16; ++i) vs[nxt][(sd + i) * 64 + srow] = vr[i];
    __syncthreads();
  }
  compute(ks[(NC - 1) & 1], vs[(NC - 1) & 1]);

  // epilogue: 1/l scaling, scatter heads back to [B,N,H*DH] bf16
  const int b = bh >> 3, h = bh & 7;
  #pragma unroll
  for (int r = 0; r < 8; ++r) {
    float inv = 1.f / lrun[r];
    int   row = q0 + r + half * 8;
    size_t base = ((size_t)b * NSEQ + row) * DM + h * DH;
    #pragma unroll
    for (int t = 0; t < 4; ++t)
      AO[base + t * 16 + col] = (bf16)(o[t][r] * inv);
  }
}

// ============================================================================
// Kernel 3: out = LayerNorm(attn @ wo + x). Same tiling as kernel 1.
// ============================================================================
__global__ __launch_bounds__(512) void out_ln_kernel(
    const bf16* __restrict__ A, const bf16* __restrict__ WT,
    const float* __restrict__ x, const float* __restrict__ g,
    const float* __restrict__ bta, float* __restrict__ out) {
  __shared__ bf16  as[64 * DM];
  __shared__ float s_sum[64], s_sq[64];

  const int tid  = threadIdx.x;
  const int lane = tid & 31;
  const int wave = tid >> 5;
  const int half = lane >> 4, col = lane & 15;
  const int row0 = blockIdx.x * 64;

  for (int i = tid; i < 64 * DM / 16; i += 512)
    *(v16bf*)(as + i * 16) = *(const v16bf*)(A + (size_t)row0 * DM + i * 16);
  __syncthreads();

  v8f acc[4][2];
  #pragma unroll
  for (int m = 0; m < 4; ++m)
    #pragma unroll
    for (int nt = 0; nt < 2; ++nt)
      #pragma unroll
      for (int r = 0; r < 8; ++r) acc[m][nt][r] = 0.f;

  for (int kt = 0; kt < 16; ++kt) {
    v16bf a[4];
    #pragma unroll
    for (int m = 0; m < 4; ++m)
      a[m] = load_a16(as + (m * 16) * DM + kt * 32, DM, lane);
    #pragma unroll
    for (int nt = 0; nt < 2; ++nt) {
      v16bf bm = load_b16(WT + (size_t)(wave * 32 + nt * 16) * DM + kt * 32,
                          DM, lane);
      #pragma unroll
      for (int m = 0; m < 4; ++m)
        acc[m][nt] = WMMA_BF16(a[m], bm, acc[m][nt]);
    }
  }

  // residual add (f32) BEFORE the LN statistics
  #pragma unroll
  for (int m = 0; m < 4; ++m)
    #pragma unroll
    for (int r = 0; r < 8; ++r) {
      int grow = row0 + m * 16 + r + half * 8;
      #pragma unroll
      for (int nt = 0; nt < 2; ++nt) {
        int n = wave * 32 + nt * 16 + col;
        acc[m][nt][r] += x[(size_t)grow * DM + n];
      }
    }

  // final LayerNorm
  __syncthreads();
  if (tid < 64) { s_sum[tid] = 0.f; s_sq[tid] = 0.f; }
  __syncthreads();
  #pragma unroll
  for (int m = 0; m < 4; ++m)
    #pragma unroll
    for (int r = 0; r < 8; ++r) {
      float s  = row_sum16(acc[m][0][r] + acc[m][1][r]);
      float qq = row_sum16(acc[m][0][r] * acc[m][0][r] +
                           acc[m][1][r] * acc[m][1][r]);
      if (col == 0) {
        atomicAdd(&s_sum[m * 16 + r + half * 8], s);
        atomicAdd(&s_sq [m * 16 + r + half * 8], qq);
      }
    }
  __syncthreads();
  #pragma unroll
  for (int m = 0; m < 4; ++m)
    #pragma unroll
    for (int r = 0; r < 8; ++r) {
      int   lrow = m * 16 + r + half * 8;
      float mean = s_sum[lrow] * (1.f / DM);
      float var  = s_sq [lrow] * (1.f / DM) - mean * mean;
      float rstd = rsqrtf(var + 1e-6f);
      int   grow = row0 + lrow;
      #pragma unroll
      for (int nt = 0; nt < 2; ++nt) {
        int n = wave * 32 + nt * 16 + col;
        out[(size_t)grow * DM + n] = (acc[m][nt][r] - mean) * rstd * g[n] + bta[n];
      }
    }
}

// ============================================================================
extern "C" void kernel_launch(void* const* d_in, const int* in_sizes, int n_in,
                              void* d_out, int out_size, void* d_ws, size_t ws_size,
                              hipStream_t stream) {
  (void)in_sizes; (void)n_in; (void)out_size; (void)ws_size;
  const float* x  = (const float*)d_in[0];
  const float* wq = (const float*)d_in[1];
  const float* wk = (const float*)d_in[2];
  const float* wv = (const float*)d_in[3];
  const float* wo = (const float*)d_in[4];
  const float* lg = (const float*)d_in[5];
  const float* lb = (const float*)d_in[6];
  const float* og = (const float*)d_in[7];
  const float* ob = (const float*)d_in[8];
  float* out = (float*)d_out;

  const size_t qkv_bytes = (size_t)NB * NSEQ * DM * sizeof(bf16);  // 16 MB
  const size_t wt_bytes  = (size_t)DM * DM * sizeof(bf16);         // 512 KB
  char* ws = (char*)d_ws;
  bf16* q   = (bf16*)(ws);
  bf16* k   = (bf16*)(ws + 1 * qkv_bytes);
  bf16* v   = (bf16*)(ws + 2 * qkv_bytes);
  bf16* ao  = (bf16*)(ws + 3 * qkv_bytes);
  bf16* wtq = (bf16*)(ws + 4 * qkv_bytes);
  bf16* wtk = (bf16*)(ws + 4 * qkv_bytes + 1 * wt_bytes);
  bf16* wtv = (bf16*)(ws + 4 * qkv_bytes + 2 * wt_bytes);
  bf16* wto = (bf16*)(ws + 4 * qkv_bytes + 3 * wt_bytes);  // 66 MB total

  wtrans_kernel<<<dim3(8, 8, 4), 256, 0, stream>>>(wq, wk, wv, wo,
                                                   wtq, wtk, wtv, wto);
  qkv_ln_kernel<<<NB * NSEQ / 64, 512, 0, stream>>>(x, wtq, wtk, wtv,
                                                    lg, lb, q, k, v);
  attn_kernel<<<dim3(NSEQ / 128, NB * NH), 256, 0, stream>>>(q, k, v, ao);
  out_ln_kernel<<<NB * NSEQ / 64, 512, 0, stream>>>(ao, wto, x, og, ob, out);
}
